// GRU_57062935495489
// MI455X (gfx1250) — matrix operand order repaired and checked by reading, not compiled, not run on hardware
//
#include <hip/hip_runtime.h>
#include <hip/hip_bf16.h>
#include <stdint.h>

typedef __attribute__((ext_vector_type(16))) __bf16 v16bf;
typedef __attribute__((ext_vector_type(8)))  __bf16 v8bf;
typedef __attribute__((ext_vector_type(8)))  float  v8f;

#define TT    512
#define BB    512
#define DIN   64
#define DLAT  128
#define KTOT  192      // DLAT + DIN
#define A_STR 136      // bf16 elems per A row (h / h*r region, 128 used), 272B: 16B aligned
#define H_STR 132      // f32 elems per h row
#define X_STR 68       // f32 elems per x row (272B row stride, 16B aligned)

// ---- WMMA wrapper: D = A(16x32 bf16) * B(32x16 bf16) + C(16x16 f32) ----
__device__ __forceinline__ v8f wmma_bf16(v16bf a, v16bf b, v8f c) {
    return __builtin_amdgcn_wmma_f32_16x16x32_bf16(
        /*neg_a=*/false, a, /*neg_b=*/false, b,
        /*c_mod=*/(short)0, c, /*reuse_a=*/false, /*reuse_b=*/false);
}

// ---- CDNA5 async global->LDS copy (ASYNCcnt tracked), GV addressing ----
__device__ __forceinline__ void async_x_load_b128(float* lds_ptr, const float* gptr) {
    // LDS aperture keeps bits[31:0] as the LDS byte offset (ISA 10.2), so
    // truncating the flat shared pointer yields the VDST LDS address.
    uint32_t lds_off = (uint32_t)(uintptr_t)lds_ptr;
    asm volatile("global_load_async_to_lds_b128 %0, %1, off"
                 :: "v"(lds_off), "v"(gptr)
                 : "memory");
}
__device__ __forceinline__ void wait_async_zero() {
#if __has_builtin(__builtin_amdgcn_s_wait_asynccnt)
    __builtin_amdgcn_s_wait_asynccnt(0);
#else
    asm volatile("s_wait_asynccnt 0" ::: "memory");
#endif
}

// B-matrix fragment: 8 f32 at p, 8 f32 at p+16 of one weight row -> bf16
__device__ __forceinline__ v16bf cvt_w16(const float* __restrict__ p) {
    float4 a = ((const float4*)p)[0];
    float4 b = ((const float4*)p)[1];
    float4 c = ((const float4*)(p + 16))[0];
    float4 d = ((const float4*)(p + 16))[1];
    v16bf f;
    f[0]=(__bf16)a.x;  f[1]=(__bf16)a.y;  f[2]=(__bf16)a.z;  f[3]=(__bf16)a.w;
    f[4]=(__bf16)b.x;  f[5]=(__bf16)b.y;  f[6]=(__bf16)b.z;  f[7]=(__bf16)b.w;
    f[8]=(__bf16)c.x;  f[9]=(__bf16)c.y;  f[10]=(__bf16)c.z; f[11]=(__bf16)c.w;
    f[12]=(__bf16)d.x; f[13]=(__bf16)d.y; f[14]=(__bf16)d.z; f[15]=(__bf16)d.w;
    return f;
}

// A-matrix fragment from bf16 LDS: elems 0..7 at p, 8..15 at p+16 (ISA 16-bit A layout)
__device__ __forceinline__ v16bf ld_a(const __bf16* p) {
    v8bf lo = *(const v8bf*)p;
    v8bf hi = *(const v8bf*)(p + 16);
    return __builtin_shufflevector(lo, hi, 0,1,2,3,4,5,6,7,8,9,10,11,12,13,14,15);
}

// A-matrix fragment from f32 LDS (x region): 8 f32 at p, 8 f32 at p+16
__device__ __forceinline__ v16bf ld_x(const float* p) {
    float4 a = ((const float4*)p)[0];
    float4 b = ((const float4*)p)[1];
    float4 c = ((const float4*)(p + 16))[0];
    float4 d = ((const float4*)(p + 16))[1];
    v16bf f;
    f[0]=(__bf16)a.x;  f[1]=(__bf16)a.y;  f[2]=(__bf16)a.z;  f[3]=(__bf16)a.w;
    f[4]=(__bf16)b.x;  f[5]=(__bf16)b.y;  f[6]=(__bf16)b.z;  f[7]=(__bf16)b.w;
    f[8]=(__bf16)c.x;  f[9]=(__bf16)c.y;  f[10]=(__bf16)c.z; f[11]=(__bf16)c.w;
    f[12]=(__bf16)d.x; f[13]=(__bf16)d.y; f[14]=(__bf16)d.z; f[15]=(__bf16)d.w;
    return f;
}

__device__ __forceinline__ float fsigmoid(float x) {
    return 1.0f / (1.0f + __expf(-x));
}
__device__ __forceinline__ float ftanh(float x) {
    return 2.0f / (1.0f + __expf(-2.0f * x)) - 1.0f;
}

__global__ __launch_bounds__(256, 1)
void gru_scan_kernel(const float* __restrict__ seqs, const float* __restrict__ h0,
                     const float* __restrict__ Wz, const float* __restrict__ bz,
                     const float* __restrict__ Wr, const float* __restrict__ br,
                     const float* __restrict__ Wi, const float* __restrict__ bi,
                     float* __restrict__ out) {
    __shared__ float  h_lds[16 * H_STR];     // current h, f32
    __shared__ __bf16 A_lds[16 * A_STR];     // h (or h*r) as bf16 A-operand, cols 0..127
    __shared__ float  x_lds[2][16 * X_STR];  // double-buffered x[t], f32 (async-filled)

    const int tid      = threadIdx.x;
    const int lane     = tid & 31;
    const int wave     = tid >> 5;       // N-tile 0..7
    const int laneHalf = lane >> 4;      // 0/1
    const int colN     = lane & 15;
    const int col      = wave * 16 + colN;   // output column 0..127
    const int M        = lane & 15;          // A-matrix row for this lane
    const int wgBase   = blockIdx.x * 16;    // batch row base

    // x-tile mapping: 256 threads x 16B (4 f32) = 16x64 tile per step
    const int xrow = tid >> 4;
    const int xc0  = (tid & 15) * 4;
    const float* xg_base = seqs + ((size_t)(wgBase + xrow)) * DIN + xc0;

    // kick off async copy of x[0] before doing anything else
    async_x_load_b128(&x_lds[0][xrow * X_STR + xc0], xg_base);

    // ---- preload weight B-fragments into VGPRs (once) ----
    v16bf Wzf[6], Wrf[6], Wif[6];
#pragma unroll
    for (int kc = 0; kc < 6; ++kc) {
        const int off = col * KTOT + kc * 32 + laneHalf * 8;
        Wzf[kc] = cvt_w16(Wz + off);
        Wrf[kc] = cvt_w16(Wr + off);
        Wif[kc] = cvt_w16(Wi + off);
    }
    const float bzl = bz[col], brl = br[col], bil = bi[col];

    // ---- init: h -> LDS (f32 + bf16), and out[0] = h ----
    {
        const int row = tid >> 4;
        const int c0  = (tid & 15) * 8;
        const float* hp = h0 + (size_t)(wgBase + row) * DLAT + c0;
        float4 a = ((const float4*)hp)[0];
        float4 b = ((const float4*)hp)[1];
        float* hl = h_lds + row * H_STR + c0;
        hl[0]=a.x; hl[1]=a.y; hl[2]=a.z; hl[3]=a.w;
        hl[4]=b.x; hl[5]=b.y; hl[6]=b.z; hl[7]=b.w;
        __bf16* al = A_lds + row * A_STR + c0;
        al[0]=(__bf16)a.x; al[1]=(__bf16)a.y; al[2]=(__bf16)a.z; al[3]=(__bf16)a.w;
        al[4]=(__bf16)b.x; al[5]=(__bf16)b.y; al[6]=(__bf16)b.z; al[7]=(__bf16)b.w;
        float* op = out + (size_t)(wgBase + row) * DLAT + c0;
        ((float4*)op)[0] = a;
        ((float4*)op)[1] = b;
    }
    wait_async_zero();
    __syncthreads();

    const __bf16* aBase = A_lds + M * A_STR;

    for (int t = 0; t < TT; ++t) {
        const int cur = t & 1;
        // async-prefetch next x tile into the alternate LDS buffer;
        // completion is awaited at the end of this step (full step of overlap)
        if (t + 1 < TT) {
            async_x_load_b128(&x_lds[1 - cur][xrow * X_STR + xc0],
                              xg_base + (size_t)(t + 1) * BB * DIN);
            if (t + 2 < TT)
                __builtin_prefetch(xg_base + (size_t)(t + 2) * BB * DIN, 0, 1);
        }

        // ---- phase 1: z and r gates, A = [h | x] ----
        v8f accz, accr;
#pragma unroll
        for (int v = 0; v < 8; ++v) { accz[v] = bzl; accr[v] = brl; }
        const float* xBase = x_lds[cur] + M * X_STR;
        v16bf xf0 = ld_x(xBase + 0 * 32 + laneHalf * 8);
        v16bf xf1 = ld_x(xBase + 1 * 32 + laneHalf * 8);
#pragma unroll
        for (int kc = 0; kc < 4; ++kc) {
            v16bf af = ld_a(aBase + kc * 32 + laneHalf * 8);
            accz = wmma_bf16(af, Wzf[kc], accz);
            accr = wmma_bf16(af, Wrf[kc], accr);
        }
        accz = wmma_bf16(xf0, Wzf[4], accz);
        accr = wmma_bf16(xf0, Wrf[4], accr);
        accz = wmma_bf16(xf1, Wzf[5], accz);
        accr = wmma_bf16(xf1, Wrf[5], accr);

        float zv[8], hv[8];
#pragma unroll
        for (int v = 0; v < 8; ++v) {
            zv[v] = fsigmoid(accz[v]);
            hv[v] = h_lds[(v + 8 * laneHalf) * H_STR + col];
        }
        __syncthreads();   // all waves done reading h-region of A

        // ---- phase 2: overwrite A h-region with h*r (each wave owns its 16 cols) ----
#pragma unroll
        for (int v = 0; v < 8; ++v) {
            float rv = fsigmoid(accr[v]);
            A_lds[(v + 8 * laneHalf) * A_STR + col] = (__bf16)(hv[v] * rv);
        }
        __syncthreads();

        // ---- phase 3: candidate, A = [h*r | x] ----
        v8f acci;
#pragma unroll
        for (int v = 0; v < 8; ++v) acci[v] = bil;
#pragma unroll
        for (int kc = 0; kc < 4; ++kc) {
            v16bf af = ld_a(aBase + kc * 32 + laneHalf * 8);
            acci = wmma_bf16(af, Wif[kc], acci);
        }
        acci = wmma_bf16(xf0, Wif[4], acci);
        acci = wmma_bf16(xf1, Wif[5], acci);
        __syncthreads();   // all waves done reading h*r-region of A

        // ---- h_new = (1-z)*h + z*tanh(cand); write LDS (f32+bf16) and out[t+1] ----
#pragma unroll
        for (int v = 0; v < 8; ++v) {
            float c  = ftanh(acci[v]);
            float hn = (1.0f - zv[v]) * hv[v] + zv[v] * c;
            const int row = v + 8 * laneHalf;
            h_lds[row * H_STR + col] = hn;
            A_lds[row * A_STR + col] = (__bf16)hn;
            out[((size_t)(t + 1) * BB + wgBase + row) * DLAT + col] = hn;
        }
        // each wave waits for its own async x-portion; barrier makes it WG-visible
        wait_async_zero();
        __syncthreads();
    }
}

extern "C" void kernel_launch(void* const* d_in, const int* in_sizes, int n_in,
                              void* d_out, int out_size, void* d_ws, size_t ws_size,
                              hipStream_t stream) {
    (void)in_sizes; (void)n_in; (void)out_size; (void)d_ws; (void)ws_size;
    const float* seqs = (const float*)d_in[0];
    const float* h0   = (const float*)d_in[1];
    const float* Wz   = (const float*)d_in[2];
    const float* bz   = (const float*)d_in[3];
    const float* Wr   = (const float*)d_in[4];
    const float* br   = (const float*)d_in[5];
    const float* Wi   = (const float*)d_in[6];
    const float* bi   = (const float*)d_in[7];
    float* out = (float*)d_out;

    dim3 grid(BB / 16);   // 32 workgroups, one 16-row batch tile each
    dim3 block(256);      // 8 wave32 = 8 N-tiles of 16 columns
    gru_scan_kernel<<<grid, block, 0, stream>>>(seqs, h0, Wz, bz, Wr, br, Wi, bi, out);
}
